// StyleNerf_18519898980813
// MI455X (gfx1250) — matrix operand order repaired and checked by reading, not compiled, not run on hardware
//
#include <hip/hip_runtime.h>

// ---------------------------------------------------------------------------
// StyleNeRF fused forward for MI455X (gfx1250), wave32 + WMMA f16.
// v3b: actB shrunk to 256-wide -> 152 KB LDS -> 2 blocks/WGP (2 waves/SIMD);
//      packed f32->f16 conversion via v_cvt_pk_rtz (result type fixed).
// ---------------------------------------------------------------------------

typedef __attribute__((ext_vector_type(16))) _Float16 v16h;
typedef __attribute__((ext_vector_type(8)))  _Float16 h8;
typedef __attribute__((ext_vector_type(2)))  __fp16   hp2;   // cvt_pkrtz result type
typedef __attribute__((ext_vector_type(8)))  float    v8f;

union AV { v16h v; h8 h[2]; };

// ---------------- workspace layout (f16 weights, padded) -------------------
constexpr size_t SZ_W0  = 256 * 64;    // layer0: 63 -> pad 64
constexpr size_t SZ_W   = 256 * 256;   // generic 256x256
constexpr size_t SZ_W5  = 256 * 320;   // skip layer: 319 -> pad 320
constexpr size_t SZ_WSG = 16 * 256;    // sigma head: 1 -> pad 16 rows
constexpr size_t SZ_WR0 = 128 * 288;   // rgb0: 283 -> pad 288
constexpr size_t SZ_WR1 = 16 * 128;    // rgb1: 3 -> pad 16 rows

constexpr size_t OW0  = 0;
constexpr size_t OW1  = OW0  + SZ_W0;
constexpr size_t OW2  = OW1  + SZ_W;
constexpr size_t OW3  = OW2  + SZ_W;
constexpr size_t OW4  = OW3  + SZ_W;
constexpr size_t OW5  = OW4  + SZ_W;
constexpr size_t OW6  = OW5  + SZ_W5;
constexpr size_t OW7  = OW6  + SZ_W;
constexpr size_t OWSG = OW7  + SZ_W;
constexpr size_t OWRM = OWSG + SZ_WSG;
constexpr size_t OWR0 = OWRM + SZ_W;
constexpr size_t OWR1 = OWR0 + SZ_WR0;
constexpr size_t W_HALVES = OWR1 + SZ_WR1;
constexpr size_t BOFF = ((W_HALVES * 2 + 255) / 256) * 256;  // byte offset of f32 biases

// bias offsets (floats): b0..b7 at i*256, then padded heads
constexpr size_t OBSG = 8 * 256;        // 2048
constexpr size_t OBRM = OBSG + 16;      // 2064
constexpr size_t OBR0 = OBRM + 256;     // 2320
constexpr size_t OBR1 = OBR0 + 128;     // 2448

// ---------------------------------------------------------------------------
__global__ void prep_w(const float* __restrict__ src, _Float16* __restrict__ dst,
                       int rows, int rows_pad, int cols, int cols_pad, int split) {
  int idx = blockIdx.x * blockDim.x + threadIdx.x;
  if (idx >= rows_pad * cols_pad) return;
  int r = idx / cols_pad, c = idx % cols_pad;
  int sc = c;
  if (split) { sc = (c < 63) ? c : (c == 63 ? -1 : c - 1); }
  float v = 0.0f;
  if (r < rows && sc >= 0 && sc < cols) v = src[(size_t)r * cols + sc];
  dst[idx] = (_Float16)v;
}

__global__ void prep_b(const float* __restrict__ src, float* __restrict__ dst,
                       int n, int npad) {
  int i = blockIdx.x * blockDim.x + threadIdx.x;
  if (i < npad) dst[i] = (i < n) ? src[i] : 0.0f;
}

// Single v_med3_f32 ReLU (avoids fmaxf's canonicalizing double-max).
__device__ __forceinline__ float relu1(float x) {
  return __builtin_amdgcn_fmed3f(x, 0.0f, 3.0e38f);
}

// ---------------------------------------------------------------------------
// Dense layer over 32 rows/wave (two 16-row M-tiles sharing each B fragment).
// A layout (ISA 7.12.2): g=lane>>4; halves 0..7 -> K=k*32+g*8+i,
//                        halves 8..15 -> K=k*32+16+g*8+i.
// B: lane%16 = N, same K order as A (k*32+g*8+i, k*32+16+g*8+i).
// D: VGPR r -> M = g*8+r ; lane%16 -> N.
// ---------------------------------------------------------------------------
template <int KSTEPS, int NTILES, bool RELU>
__device__ __forceinline__ void ffn_layer(
    const _Float16* __restrict__ W, int kpad, const float* __restrict__ bias,
    const _Float16* in, int in_stride, int in_col,
    _Float16* out, int out_stride, int out_col) {
  const int lane = threadIdx.x & 31;
  const int wave = threadIdx.x >> 5;
  const int g = lane >> 4, ln = lane & 15;
  const int row0 = wave * 32 + ln;   // M-tile 0
  const int row1 = row0 + 16;        // M-tile 1

  AV A0[KSTEPS], A1[KSTEPS];
  const _Float16* ap0 = in + (size_t)row0 * in_stride + in_col + g * 8;
  const _Float16* ap1 = in + (size_t)row1 * in_stride + in_col + g * 8;
#pragma unroll
  for (int k = 0; k < KSTEPS; ++k) {
    A0[k].h[0] = *(const h8*)(ap0 + k * 32);
    A0[k].h[1] = *(const h8*)(ap0 + k * 32 + 16);
    A1[k].h[0] = *(const h8*)(ap1 + k * 32);
    A1[k].h[1] = *(const h8*)(ap1 + k * 32 + 16);
  }

#pragma unroll
  for (int nt = 0; nt < NTILES; ++nt) {
    const float b = bias[nt * 16 + ln];
    v8f acc0, acc1;
#pragma unroll
    for (int r = 0; r < 8; ++r) { acc0[r] = b; acc1[r] = b; }

    const _Float16* wp = W + (size_t)(nt * 16 + ln) * kpad + g * 8;
    if (nt + 1 < NTILES)
      __builtin_prefetch(W + (size_t)((nt + 1) * 16 + ln) * kpad + g * 16, 0, 3);
#pragma unroll
    for (int k = 0; k < KSTEPS; ++k) {
      AV B;
      B.h[0] = *(const h8*)(wp + k * 32);
      B.h[1] = *(const h8*)(wp + k * 32 + 16);
      acc0 = __builtin_amdgcn_wmma_f32_16x16x32_f16(false, A0[k].v, false, B.v,
                                                    (short)0, acc0, false, false);
      acc1 = __builtin_amdgcn_wmma_f32_16x16x32_f16(false, A1[k].v, false, B.v,
                                                    (short)0, acc1, false, false);
    }
    _Float16* op0 = out + (size_t)(wave * 32 + g * 8) * out_stride + out_col + nt * 16 + ln;
    _Float16* op1 = op0 + (size_t)16 * out_stride;
#pragma unroll
    for (int r = 0; r < 8; r += 2) {
      float x0a = acc0[r], x0b = acc0[r + 1];
      float x1a = acc1[r], x1b = acc1[r + 1];
      if (RELU) {
        x0a = relu1(x0a); x0b = relu1(x0b);
        x1a = relu1(x1a); x1b = relu1(x1b);
      }
      hp2 p0 = __builtin_amdgcn_cvt_pkrtz(x0a, x0b);   // one v_cvt_pk per pair
      hp2 p1 = __builtin_amdgcn_cvt_pkrtz(x1a, x1b);
      op0[(size_t)r * out_stride] = (_Float16)p0[0];
      op0[(size_t)(r + 1) * out_stride] = (_Float16)p0[1];
      op1[(size_t)r * out_stride] = (_Float16)p1[0];
      op1[(size_t)(r + 1) * out_stride] = (_Float16)p1[1];
    }
  }
}

// Single-N-tile head over 32 rows/wave; returns both raw accumulator frags.
struct Acc2 { v8f a0, a1; };
template <int KSTEPS>
__device__ __forceinline__ Acc2 ffn_head(
    const _Float16* __restrict__ W, int kpad, const float* __restrict__ bias,
    const _Float16* in, int in_stride, int in_col) {
  const int lane = threadIdx.x & 31;
  const int wave = threadIdx.x >> 5;
  const int g = lane >> 4, ln = lane & 15;
  const int row0 = wave * 32 + ln;

  const _Float16* ap0 = in + (size_t)row0 * in_stride + in_col + g * 8;
  const _Float16* ap1 = ap0 + (size_t)16 * in_stride;
  const _Float16* wp = W + (size_t)ln * kpad + g * 8;
  const float b = bias[ln];
  Acc2 acc;
#pragma unroll
  for (int r = 0; r < 8; ++r) { acc.a0[r] = b; acc.a1[r] = b; }
#pragma unroll
  for (int k = 0; k < KSTEPS; ++k) {
    AV A0, A1, B;
    A0.h[0] = *(const h8*)(ap0 + k * 32);
    A0.h[1] = *(const h8*)(ap0 + k * 32 + 16);
    A1.h[0] = *(const h8*)(ap1 + k * 32);
    A1.h[1] = *(const h8*)(ap1 + k * 32 + 16);
    B.h[0] = *(const h8*)(wp + k * 32);
    B.h[1] = *(const h8*)(wp + k * 32 + 16);
    acc.a0 = __builtin_amdgcn_wmma_f32_16x16x32_f16(false, A0.v, false, B.v,
                                                    (short)0, acc.a0, false, false);
    acc.a1 = __builtin_amdgcn_wmma_f32_16x16x32_f16(false, A1.v, false, B.v,
                                                    (short)0, acc.a1, false, false);
  }
  return acc;
}

// ---------------------------------------------------------------------------
// Fused MLP: 128 points per block, 4 waves, each wave owns 32 rows.
// LDS: actA[128][320] + actB[128][256] + dbuf[128][32] f16 = 152 KB
//      -> 2 blocks per WGP (2 waves per SIMD32).
// ---------------------------------------------------------------------------
__global__ void __launch_bounds__(128)
nerf_fused_kernel(const float* __restrict__ pts, const float* __restrict__ dirs,
                  const _Float16* __restrict__ wb, const float* __restrict__ bb,
                  float* __restrict__ rgb, float* __restrict__ sigma,
                  float* __restrict__ demb) {
  extern __shared__ _Float16 smem[];
  _Float16* actA = smem;               // [128][320]
  _Float16* actB = smem + 128 * 320;   // [128][256]
  _Float16* dbuf = actB + 128 * 256;   // [128][32]

  const int t = threadIdx.x;
  const int tile = blockIdx.x * 128;

  // ---- positional encodings: thread t handles row t (wave-local) ----
  {
    const size_t p = (size_t)(tile + t);
    {
      const float x = pts[p * 3], y = pts[p * 3 + 1], z = pts[p * 3 + 2];
      _Float16* ar = actA + (size_t)t * 320;
      ar[0] = (_Float16)x; ar[1] = (_Float16)y; ar[2] = (_Float16)z;
      ar[63] = (_Float16)0.0f;
      float f = 1.0f;
#pragma unroll
      for (int i = 0; i < 10; ++i) {
        _Float16* o = ar + 3 + i * 6;
        o[0] = (_Float16)__sinf(f * x); o[1] = (_Float16)__sinf(f * y);
        o[2] = (_Float16)__sinf(f * z);
        o[3] = (_Float16)__cosf(f * x); o[4] = (_Float16)__cosf(f * y);
        o[5] = (_Float16)__cosf(f * z);
        f *= 2.0f;
      }
    }
    {
      const float x = dirs[p * 3], y = dirs[p * 3 + 1], z = dirs[p * 3 + 2];
      _Float16* dr = dbuf + (size_t)t * 32;
      float* od = demb + p * 27;
      dr[0] = (_Float16)x; dr[1] = (_Float16)y; dr[2] = (_Float16)z;
      od[0] = x; od[1] = y; od[2] = z;
      float f = 1.0f;
#pragma unroll
      for (int i = 0; i < 4; ++i) {
        const float s0 = __sinf(f * x), s1 = __sinf(f * y), s2 = __sinf(f * z);
        const float c0 = __cosf(f * x), c1 = __cosf(f * y), c2 = __cosf(f * z);
        _Float16* o = dr + 3 + i * 6;
        float* oo = od + 3 + i * 6;
        o[0] = (_Float16)s0; o[1] = (_Float16)s1; o[2] = (_Float16)s2;
        o[3] = (_Float16)c0; o[4] = (_Float16)c1; o[5] = (_Float16)c2;
        oo[0] = s0; oo[1] = s1; oo[2] = s2;
        oo[3] = c0; oo[4] = c1; oo[5] = c2;
        f *= 2.0f;
      }
#pragma unroll
      for (int j = 27; j < 32; ++j) dr[j] = (_Float16)0.0f;
    }
  }
  __syncthreads();

  // ---- base MLP (p stays in actA cols 0..63 until the skip layer) ----
  ffn_layer<2, 16, true>(wb + OW0, 64, bb + 0 * 256, actA, 320, 0, actA, 320, 64);
  ffn_layer<8, 16, true>(wb + OW1, 256, bb + 1 * 256, actA, 320, 64, actB, 256, 0);
  ffn_layer<8, 16, true>(wb + OW2, 256, bb + 2 * 256, actB, 256, 0, actA, 320, 64);
  ffn_layer<8, 16, true>(wb + OW3, 256, bb + 3 * 256, actA, 320, 64, actB, 256, 0);
  ffn_layer<8, 16, true>(wb + OW4, 256, bb + 4 * 256, actB, 256, 0, actA, 320, 64);
  // skip layer: input = [p | base4] = actA cols 0..319, K=320
  ffn_layer<10, 16, true>(wb + OW5, 320, bb + 5 * 256, actA, 320, 0, actB, 256, 0);
  ffn_layer<8, 16, true>(wb + OW6, 256, bb + 6 * 256, actB, 256, 0, actA, 320, 64);
  ffn_layer<8, 16, true>(wb + OW7, 256, bb + 7 * 256, actA, 320, 64, actB, 256, 0);

  const int lane = t & 31, wave = t >> 5;
  const int g = lane >> 4, ln = lane & 15;

  // ---- sigma head (SIGMA_MUL == 0 -> raw linear), only N==0 is real ----
  {
    Acc2 s = ffn_head<8>(wb + OWSG, 256, bb + OBSG, actB, 256, 0);
    if (ln == 0) {
      const int rb = tile + wave * 32 + g * 8;
#pragma unroll
      for (int r = 0; r < 8; ++r) {
        sigma[rb + r] = s.a0[r];
        sigma[rb + 16 + r] = s.a1[r];
      }
    }
  }

  // ---- base_remap: actB.0 -> actA cols 0..255 ----
  ffn_layer<8, 16, true>(wb + OWRM, 256, bb + OBRM, actB, 256, 0, actA, 320, 0);

  // ---- splice dir embedding into actA cols 256..287 (row t, wave-local) ----
  {
    const h8* s = (const h8*)(dbuf + (size_t)t * 32);
    h8* d = (h8*)(actA + (size_t)t * 320 + 256);
    d[0] = s[0]; d[1] = s[1]; d[2] = s[2]; d[3] = s[3];
  }

  // ---- rgb0: [remap | d] (K=288) -> 128 ch in actB cols 0..127 ----
  ffn_layer<9, 8, true>(wb + OWR0, 288, bb + OBR0, actA, 320, 0, actB, 256, 0);

  // ---- rgb1 head: 128 -> 3 (padded 16), sigmoid, lanes N<3 are real ----
  {
    Acc2 c = ffn_head<4>(wb + OWR1, 128, bb + OBR1, actB, 256, 0);
    if (ln < 3) {
#pragma unroll
      for (int r = 0; r < 8; ++r) {
        const size_t p0 = (size_t)(tile + wave * 32 + g * 8 + r);
        const size_t p1 = p0 + 16;
        rgb[p0 * 3 + ln] = 1.0f / (1.0f + __expf(-c.a0[r]));
        rgb[p1 * 3 + ln] = 1.0f / (1.0f + __expf(-c.a1[r]));
      }
    }
  }
}

// ---------------------------------------------------------------------------
extern "C" void kernel_launch(void* const* d_in, const int* in_sizes, int n_in,
                              void* d_out, int out_size, void* d_ws, size_t ws_size,
                              hipStream_t stream) {
  const float* pts  = (const float*)d_in[0];
  const float* dirs = (const float*)d_in[1];
  const float* Wb[8];
  const float* bbv[8];
  for (int i = 0; i < 8; ++i) {
    Wb[i]  = (const float*)d_in[2 + 2 * i];
    bbv[i] = (const float*)d_in[3 + 2 * i];
  }
  const float* Wsig = (const float*)d_in[18];
  const float* bsig = (const float*)d_in[19];
  const float* Wrm  = (const float*)d_in[20];
  const float* brm  = (const float*)d_in[21];
  const float* Wr0  = (const float*)d_in[22];
  const float* br0  = (const float*)d_in[23];
  const float* Wr1  = (const float*)d_in[24];
  const float* br1  = (const float*)d_in[25];

  _Float16* wbase = (_Float16*)d_ws;
  float* bbase = (float*)((char*)d_ws + BOFF);

  auto launch_w = [&](const float* src, size_t off, int rows, int rows_pad,
                      int cols, int cols_pad, int split) {
    const int n = rows_pad * cols_pad;
    prep_w<<<(n + 255) / 256, 256, 0, stream>>>(src, wbase + off, rows, rows_pad,
                                                cols, cols_pad, split);
  };
  launch_w(Wb[0], OW0, 256, 256, 63, 64, 0);
  launch_w(Wb[1], OW1, 256, 256, 256, 256, 0);
  launch_w(Wb[2], OW2, 256, 256, 256, 256, 0);
  launch_w(Wb[3], OW3, 256, 256, 256, 256, 0);
  launch_w(Wb[4], OW4, 256, 256, 256, 256, 0);
  launch_w(Wb[5], OW5, 256, 256, 319, 320, 1);  // skip-concat remap
  launch_w(Wb[6], OW6, 256, 256, 256, 256, 0);
  launch_w(Wb[7], OW7, 256, 256, 256, 256, 0);
  launch_w(Wsig, OWSG, 1, 16, 256, 256, 0);
  launch_w(Wrm, OWRM, 256, 256, 256, 256, 0);
  launch_w(Wr0, OWR0, 128, 128, 283, 288, 0);
  launch_w(Wr1, OWR1, 3, 16, 128, 128, 0);

  auto launch_b = [&](const float* src, size_t off, int n, int npad) {
    prep_b<<<(npad + 255) / 256, 256, 0, stream>>>(src, bbase + off, n, npad);
  };
  for (int i = 0; i < 8; ++i) launch_b(bbv[i], (size_t)i * 256, 256, 256);
  launch_b(bsig, OBSG, 1, 16);
  launch_b(brm, OBRM, 256, 256);
  launch_b(br0, OBR0, 128, 128);
  launch_b(br1, OBR1, 3, 16);

  const int N = in_sizes[0] / 3;  // 262144
  float* out = (float*)d_out;
  float* rgb_out = out;                       // [N,3]
  float* sigma_out = out + (size_t)N * 3;     // [N]
  float* demb_out = out + (size_t)N * 4;      // [N,27]

  // actA 128x320 + actB 128x256 + dbuf 128x32, f16 = 152 KB
  const size_t shmem = (size_t)(128 * 320 + 128 * 256 + 128 * 32) * sizeof(_Float16);
  nerf_fused_kernel<<<N / 128, 128, shmem, stream>>>(pts, dirs, wbase, bbase,
                                                     rgb_out, sigma_out, demb_out);
}
